// MultiModel_59923383714397
// MI455X (gfx1250) — matrix operand order repaired and checked by reading, not compiled, not run on hardware
//
#include <hip/hip_runtime.h>
#include <hip/hip_bf16.h>

// ---------- types for WMMA bf16 path (gfx1250) ----------
typedef __attribute__((ext_vector_type(16))) __bf16 v16bf;
typedef __attribute__((ext_vector_type(8)))  float  v8f;

union FragU {
    uint4 u4[2];
    v16bf v;
};

__device__ __forceinline__ unsigned f2bf_bits(float f) {
    unsigned u = __builtin_bit_cast(unsigned, f);
    return (u + 0x7FFFu + ((u >> 16) & 1u)) >> 16;   // RNE bf16
}
__device__ __forceinline__ unsigned short f2bf(float f) {
    return (unsigned short)f2bf_bits(f);
}
__device__ __forceinline__ float sigm(float x) {
    return 1.0f / (1.0f + __expf(-x));
}
// branch-free tanh: 1 - 2/(e^{2x}+1); saturates to +-1 for large |x|
__device__ __forceinline__ float ftanh(float x) {
    return 1.0f - 2.0f / (__expf(2.0f * x) + 1.0f);
}

// Model dims
#define BB   64
#define TT   256
#define DIN  800      // 2*(Dc+Db)
#define HID  512
#define G4H  2048     // 4*H
#define KCMB 1024     // HID + H (fused [x;h] GEMM K)

// =====================================================================
// 1) Embedding gather + concat -> bf16 activation rows [T*B, 800]
//    row m = t*64 + b  (so projection output is directly [T,B,HID])
// =====================================================================
__global__ void embed_kernel(const int* __restrict__ ci,  const int* __restrict__ bli,
                             const int* __restrict__ bri, const int* __restrict__ sci,
                             const int* __restrict__ sbli,const int* __restrict__ sbri,
                             const float* __restrict__ ct,  const float* __restrict__ sct,
                             const float* __restrict__ bt,  const float* __restrict__ sbt,
                             unsigned short* __restrict__ leftC,
                             unsigned short* __restrict__ rightC)
{
    int pos = blockIdx.x;            // b*T + t
    int b = pos >> 8, t = pos & 255;
    int m = t * BB + b;
    const float* pch  = ct  + (size_t)ci[pos]   * 200;
    const float* psch = sct + (size_t)sci[pos]  * 200;
    const float* pbl  = bt  + (size_t)bli[pos]  * 200;
    const float* pbr  = bt  + (size_t)bri[pos]  * 200;
    const float* psbl = sbt + (size_t)sbli[pos] * 200;
    const float* psbr = sbt + (size_t)sbri[pos] * 200;
    unsigned short* lo = leftC  + (size_t)m * DIN;
    unsigned short* ro = rightC + (size_t)m * DIN;
    for (int e = threadIdx.x; e < DIN; e += 256) {
        float lv, rv;
        if      (e < 200) { lv = rv = pch[e]; }
        else if (e < 400) { lv = rv = psch[e - 200]; }
        else if (e < 600) { lv = pbl[e - 400];  rv = pbr[e - 400]; }
        else              { lv = psbl[e - 600]; rv = psbr[e - 600]; }
        lo[e] = f2bf(lv);
        ro[e] = f2bf(rv);
    }
}

// =====================================================================
// 2) Pack a (transposed) weight matrix B[k][n] into WMMA B-fragment
//    layout: dword index ((kb*NB+nb)*32 + lane)*8 + j holds the bf16
//    pair B[kb*32 + (j/4)*16 + half*8 + 2(j%4) .. +1][nb*16 + lane%16], half=lane/16.
//    elem(k,n) = k<ksplit ? A0[n*ld0+k] : A1[n*ld1+(k-ksplit)]
// =====================================================================
__global__ void pack_b_kernel(const float* __restrict__ A0, const float* __restrict__ A1,
                              int N, int ksplit, int ld0, int ld1,
                              unsigned* __restrict__ dst, int ndw)
{
    int tid = blockIdx.x * blockDim.x + threadIdx.x;
    if (tid >= ndw) return;
    int j    = tid & 7;
    int lane = (tid >> 3) & 31;
    int fb   = tid >> 8;
    int NB   = N >> 4;
    int nb   = fb % NB, kb = fb / NB;
    int ln   = lane & 15, half = lane >> 4;
    int k = kb * 32 + (j >> 2) * 16 + half * 8 + 2 * (j & 3);
    int n = nb * 16 + ln;
    float e0 = (k < ksplit) ? A0[(size_t)n * ld0 + k]
                            : A1[(size_t)n * ld1 + (k - ksplit)];
    float e1 = (k + 1 < ksplit) ? A0[(size_t)n * ld0 + k + 1]
                                : A1[(size_t)n * ld1 + (k + 1 - ksplit)];
    dst[tid] = f2bf_bits(e0) | (f2bf_bits(e1) << 16);
}

__global__ void bias_kernel(const float* __restrict__ a, const float* __restrict__ b,
                            float* __restrict__ dst, int n)
{
    int i = blockIdx.x * blockDim.x + threadIdx.x;
    if (i < n) dst[i] = a[i] + b[i];
}

// =====================================================================
// 3) Projection GEMM: X = tanh(C @ Wlin^T + b_lin) -> bf16 [T*B, 512]
//    M=16384, N=512, K=800.  256 thr = 8 waves; tile 128x64; 4 acc/wave.
//    B fragments double-buffered in pairs so WMMA overlaps the partner
//    fragment's in-flight loads (partial s_wait_loadcnt).
// =====================================================================
__global__ void __launch_bounds__(256) proj_kernel(
    const unsigned short* __restrict__ leftC, const unsigned short* __restrict__ rightC,
    const unsigned* __restrict__ WlinP, const float* __restrict__ b_lin,
    unsigned short* __restrict__ Xl, unsigned short* __restrict__ Xr)
{
    const unsigned short* A = blockIdx.z ? rightC : leftC;
    unsigned short*       X = blockIdx.z ? Xr     : Xl;
    int tid = threadIdx.x, lane = tid & 31, wave = tid >> 5;
    int ln = lane & 15, half = lane >> 4;
    int m0 = blockIdx.x * 128 + wave * 16;
    int n0 = blockIdx.y * 64;

    v8f acc[4] = {};
    const unsigned short* arow = A + (size_t)(m0 + ln) * DIN;
    // one base per k-step; 4 fragments at immediate offsets i*1024 B
    const uint4* bp4 = (const uint4*)(WlinP + (((size_t)(blockIdx.y * 4) * 32 + lane) << 3));

#pragma unroll 1
    for (int kb = 0; kb < DIN / 32; ++kb) {           // 25 k-steps, keep rolled
        FragU a;
        const uint4* ap = (const uint4*)(arow + kb * 32 + half * 8);
        a.u4[0] = ap[0];
        a.u4[1] = ap[2];
#pragma unroll
        for (int i = 0; i < 4; i += 2) {
            FragU b0, b1;
            b0.u4[0] = bp4[i * 64];                   // +i*1024 bytes (immediate)
            b0.u4[1] = bp4[i * 64 + 1];
            b1.u4[0] = bp4[i * 64 + 64];
            b1.u4[1] = bp4[i * 64 + 65];
            acc[i] = __builtin_amdgcn_wmma_f32_16x16x32_bf16(
                false, a.v, false, b0.v, (short)0, acc[i], false, false);
            acc[i + 1] = __builtin_amdgcn_wmma_f32_16x16x32_bf16(
                false, a.v, false, b1.v, (short)0, acc[i + 1], false, false);
        }
        bp4 += 32 * 16;                               // per k-step: 512 uint4 = 2048 dwords (8 tiles * 256 dwords)
    }
#pragma unroll
    for (int i = 0; i < 4; ++i) {
        int n = n0 + i * 16 + ln;
        float bv = b_lin[n];
#pragma unroll
        for (int r = 0; r < 8; ++r) {
            int m = m0 + r + 8 * half;                // C/D layout: M = r + 8*half
            X[(size_t)m * HID + n] = f2bf(ftanh(acc[i][r] + bv));
        }
    }
}

// =====================================================================
// 4) Recurrent LSTM, one 32-wave workgroup per direction.
//    Per step: gates[64,2048] = [x_t ; h](64x1024) @ Wcomb(1024x2048)+bias
//    then pointwise cell; h kept in LDS as bf16 for the next step's A.
//    K-loop split into global-X half and LDS-h half (branch-free bodies),
//    rolled k-loops (pinned accumulators), paired double-buffered B frags.
// =====================================================================
__global__ void __launch_bounds__(1024) lstm_kernel(
    const unsigned short* __restrict__ Xl, const unsigned short* __restrict__ Xr,
    const unsigned* __restrict__ WPl, const unsigned* __restrict__ WPr,
    const float* __restrict__ biasl, const float* __restrict__ biasr,
    const float* __restrict__ h0l, const float* __restrict__ c0l,
    const float* __restrict__ h0r, const float* __restrict__ c0r,
    float* __restrict__ csl, float* __restrict__ csr,
    float* __restrict__ gbl, float* __restrict__ gbr,
    float* __restrict__ out)
{
    const int dir = blockIdx.x;
    const unsigned short* X   = dir ? Xr   : Xl;
    const unsigned*       WP  = dir ? WPr  : WPl;
    const float*          bia = dir ? biasr: biasl;
    const float*          h0  = dir ? h0r  : h0l;
    const float*          c0  = dir ? c0r  : c0l;
    float*                cs  = dir ? csr  : csl;
    float*                gb  = dir ? gbr  : gbl;

    __shared__ unsigned short Hs[BB * HID];           // 64 KB bf16 h state

    const int tid = threadIdx.x;
    for (int i = 0; i < 32; ++i) {
        int e = tid + i * 1024;
        Hs[e] = f2bf(h0[e]);
        cs[e] = c0[e];
    }
    __syncthreads();

    const int lane = tid & 31, wave = tid >> 5;
    const int ln = lane & 15, half = lane >> 4;
    const int m0 = (wave & 3) * 16;                   // row tile (M=64 -> 4 tiles)
    const int cbase = (wave >> 2) * 16;               // 16 col tiles per wave

    const unsigned* wbase0 = WP + (((size_t)cbase * 32 + lane) << 3);
    const unsigned short* hrow = &Hs[(m0 + ln) * HID];

#pragma unroll 1
    for (int s = 0; s < TT; ++s) {
        const int t = dir ? (TT - 1 - s) : s;
        const unsigned short* xrow = X + (size_t)(t * BB + m0 + ln) * HID;

        // prefetch next step's x row while this step computes
        if (s + 1 < TT) {
            const int tn = dir ? (t - 1) : (t + 1);
            __builtin_prefetch(X + (size_t)(tn * BB + m0 + ln) * HID, 0, 1);
        }

        v8f acc[16] = {};
        const uint4* bp4 = (const uint4*)wbase0;

        // ---- K half 1: x_t from global (L2-hot) ----
#pragma unroll 1
        for (int kb = 0; kb < 16; ++kb) {
            FragU a;
            const uint4* ap = (const uint4*)(xrow + kb * 32 + half * 8);
            a.u4[0] = ap[0];
            a.u4[1] = ap[2];
#pragma unroll
            for (int i = 0; i < 16; i += 2) {
                FragU b0, b1;
                b0.u4[0] = bp4[i * 64];               // +i*1024 B immediate
                b0.u4[1] = bp4[i * 64 + 1];
                b1.u4[0] = bp4[i * 64 + 64];
                b1.u4[1] = bp4[i * 64 + 65];
                acc[i] = __builtin_amdgcn_wmma_f32_16x16x32_bf16(
                    false, a.v, false, b0.v, (short)0, acc[i], false, false);
                acc[i + 1] = __builtin_amdgcn_wmma_f32_16x16x32_bf16(
                    false, a.v, false, b1.v, (short)0, acc[i + 1], false, false);
            }
            bp4 += 128 * 64;                          // next k-step: 128 tiles * 256 dwords
        }
        // ---- K half 2: h from LDS (ds_load_b128) ----
#pragma unroll 1
        for (int kb = 0; kb < 16; ++kb) {
            FragU a;
            const uint4* ap = (const uint4*)(hrow + kb * 32 + half * 8);
            a.u4[0] = ap[0];
            a.u4[1] = ap[2];
#pragma unroll
            for (int i = 0; i < 16; i += 2) {
                FragU b0, b1;
                b0.u4[0] = bp4[i * 64];
                b0.u4[1] = bp4[i * 64 + 1];
                b1.u4[0] = bp4[i * 64 + 64];
                b1.u4[1] = bp4[i * 64 + 65];
                acc[i] = __builtin_amdgcn_wmma_f32_16x16x32_bf16(
                    false, a.v, false, b0.v, (short)0, acc[i], false, false);
                acc[i + 1] = __builtin_amdgcn_wmma_f32_16x16x32_bf16(
                    false, a.v, false, b1.v, (short)0, acc[i + 1], false, false);
            }
            bp4 += 128 * 64;
        }

        // write gate pre-activations (+bias) to L2-resident scratch
#pragma unroll
        for (int i = 0; i < 16; ++i) {
            int n = (cbase + i) * 16 + ln;
            float bv = bia[n];
#pragma unroll
            for (int r = 0; r < 8; ++r)
                gb[(m0 + r + 8 * half) * G4H + n] = acc[i][r] + bv;
        }
        __threadfence_block();
        __syncthreads();

        // pointwise LSTM cell: each thread owns 32 fixed (b,n) elements
        for (int i = 0; i < 32; ++i) {
            int e = tid + i * 1024;
            int b = e >> 9, n = e & 511;
            const float* g = gb + b * G4H;
            float ig = sigm(g[n]);
            float fg = sigm(g[512 + n]);
            float gg = ftanh(g[1024 + n]);
            float og = sigm(g[1536 + n]);
            float c = fg * cs[e] + ig * gg;
            cs[e] = c;
            float h = og * ftanh(c);
            out[((size_t)(b * TT + t)) * (2 * HID) + dir * HID + n] = h;
            Hs[e] = f2bf(h);
        }
        __threadfence_block();
        __syncthreads();
    }
}

// =====================================================================
// launcher
// =====================================================================
extern "C" void kernel_launch(void* const* d_in, const int* in_sizes, int n_in,
                              void* d_out, int out_size, void* d_ws, size_t ws_size,
                              hipStream_t stream)
{
    const int*   char_ids   = (const int*)d_in[0];
    const int*   bl_ids     = (const int*)d_in[1];
    const int*   br_ids     = (const int*)d_in[2];
    const int*   sc_ids     = (const int*)d_in[3];
    const int*   sbl_ids    = (const int*)d_in[4];
    const int*   sbr_ids    = (const int*)d_in[5];
    const float* char_tab   = (const float*)d_in[6];
    const float* schar_tab  = (const float*)d_in[7];
    const float* bichar_tab = (const float*)d_in[8];
    const float* sbichar_tab= (const float*)d_in[9];
    const float* W_lin      = (const float*)d_in[10];
    const float* b_lin      = (const float*)d_in[11];
    const float* W_ih_l     = (const float*)d_in[12];
    const float* W_hh_l     = (const float*)d_in[13];
    const float* b_ih_l     = (const float*)d_in[14];
    const float* b_hh_l     = (const float*)d_in[15];
    const float* W_ih_r     = (const float*)d_in[16];
    const float* W_hh_r     = (const float*)d_in[17];
    const float* b_ih_r     = (const float*)d_in[18];
    const float* b_hh_r     = (const float*)d_in[19];
    const float* h0_l       = (const float*)d_in[20];
    const float* c0_l       = (const float*)d_in[21];
    const float* h0_r       = (const float*)d_in[22];
    const float* c0_r       = (const float*)d_in[23];
    float* out = (float*)d_out;

    // ---- workspace carving (256 B aligned) ----
    char* ws = (char*)d_ws;
    size_t off = 0;
    auto carve = [&](size_t bytes) -> char* {
        char* p = ws + off;
        off += (bytes + 255) & ~(size_t)255;
        return p;
    };
    const size_t M = (size_t)TT * BB;                         // 16384 rows
    unsigned short* leftC  = (unsigned short*)carve(M * DIN * 2);
    unsigned short* rightC = (unsigned short*)carve(M * DIN * 2);
    unsigned*       WlinP  = (unsigned*)carve((size_t)DIN * HID / 2 * 4);
    unsigned*       WPl    = (unsigned*)carve((size_t)KCMB * G4H / 2 * 4);
    unsigned*       WPr    = (unsigned*)carve((size_t)KCMB * G4H / 2 * 4);
    unsigned short* Xl     = (unsigned short*)carve(M * HID * 2);
    unsigned short* Xr     = (unsigned short*)carve(M * HID * 2);
    float*          biasl  = (float*)carve(G4H * 4);
    float*          biasr  = (float*)carve(G4H * 4);
    float*          gbl    = (float*)carve((size_t)BB * G4H * 4);
    float*          gbr    = (float*)carve((size_t)BB * G4H * 4);
    float*          csl    = (float*)carve((size_t)BB * HID * 4);
    float*          csr    = (float*)carve((size_t)BB * HID * 4);

    // 1) embedding gather + bf16 concat
    embed_kernel<<<BB * TT, 256, 0, stream>>>(
        char_ids, bl_ids, br_ids, sc_ids, sbl_ids, sbr_ids,
        char_tab, schar_tab, bichar_tab, sbichar_tab, leftC, rightC);

    // 2) weight packing into WMMA B-fragment layout
    {
        int ndw = DIN * HID / 2;                              // 204800
        pack_b_kernel<<<(ndw + 255) / 256, 256, 0, stream>>>(
            W_lin, W_lin, HID, DIN, DIN, DIN, WlinP, ndw);
    }
    {
        int ndw = KCMB * G4H / 2;                             // 1048576
        pack_b_kernel<<<(ndw + 255) / 256, 256, 0, stream>>>(
            W_ih_l, W_hh_l, G4H, HID, HID, HID, WPl, ndw);
        pack_b_kernel<<<(ndw + 255) / 256, 256, 0, stream>>>(
            W_ih_r, W_hh_r, G4H, HID, HID, HID, WPr, ndw);
    }
    bias_kernel<<<(G4H + 255) / 256, 256, 0, stream>>>(b_ih_l, b_hh_l, biasl, G4H);
    bias_kernel<<<(G4H + 255) / 256, 256, 0, stream>>>(b_ih_r, b_hh_r, biasr, G4H);

    // 3) projection GEMM + tanh (WMMA bf16), both directions via grid.z
    proj_kernel<<<dim3((unsigned)(M / 128), HID / 64, 2), 256, 0, stream>>>(
        leftC, rightC, WlinP, b_lin, Xl, Xr);

    // 4) recurrent bi-LSTM (one persistent 32-wave WG per direction)
    lstm_kernel<<<2, 1024, 0, stream>>>(
        Xl, Xr, WPl, WPr, biasl, biasr,
        h0_l, c0_l, h0_r, c0_r, csl, csr, gbl, gbr, out);

    (void)in_sizes; (void)n_in; (void)out_size; (void)ws_size;
}
